// GraphPatcherLayer_55379308315428
// MI455X (gfx1250) — matrix operand order repaired and checked by reading, not compiled, not run on hardware
//
#include <hip/hip_runtime.h>

typedef __attribute__((ext_vector_type(16))) _Float16 v16h;
typedef __attribute__((ext_vector_type(8)))  float    v8f;
typedef __attribute__((ext_vector_type(4)))  int      v4i;

#define GLOBAL_AS __attribute__((address_space(1)))
#define LDS_AS    __attribute__((address_space(3)))

#define HID 64
#define HEADS 4
#define NEG_SLOPE 0.2f
#define LN_EPS 1e-5f

// Detect gfx1250 async global->LDS builtins (ASYNCcnt path)
#if defined(__has_builtin)
#  if __has_builtin(__builtin_amdgcn_global_load_async_to_lds_b128)
#    define USE_ASYNC_LDS 1
#  endif
#endif
#ifndef USE_ASYNC_LDS
#  define USE_ASYNC_LDS 0
#endif

// ---------------------------------------------------------------------------
// Pack a row-major f32 weight W[K, NC] into WMMA B-fragment order (f16).
// B 32x16 layout (wave32): lanes 0-15 hold K=0..7 (halves 0..7), K=16..23 (halves 8..15); lanes
// 16-31 hold K+8; N = lane&15.  Pack index: ((tile*NQ+q)*32+lane)*16+h
// so the GEMM reads one contiguous 32B v16h per lane.
// ---------------------------------------------------------------------------
__global__ void pack_b_kernel(const float* __restrict__ W, _Float16* __restrict__ out,
                              int K, int NC) {
  int t = blockIdx.x * blockDim.x + threadIdx.x;
  int NQ = K >> 5;
  int NT = NC >> 4;
  int total = NT * NQ * 32;
  if (t >= total) return;
  int lane = t & 31;
  int q    = (t >> 5) % NQ;
  int tile = t / (32 * NQ);
  int n = tile * 16 + (lane & 15);
  int g = lane >> 4;
  _Float16* dst = out + (size_t)t * 16;
  #pragma unroll 1
  for (int h = 0; h < 16; ++h) {
    int k = q * 32 + 8 * g + (h & 7) + ((h >> 3) << 4);
    dst[h] = (_Float16)W[(size_t)k * NC + n];
  }
}

// ---------------------------------------------------------------------------
// WMMA GEMM: C[N, NC] = A[N, K] @ B(pre-packed f16) (+bias) (+relu)
// One block = NC/16 waves; each wave owns one 16x16 N-tile; 16 A-rows staged
// in padded LDS (via GLOBAL_LOAD_ASYNC_TO_LDS_B128 when available) and
// converted to f16 fragments per the ISA layout:
// lanes 0-15: halves 0..7 -> K 0..7, halves 8..15 -> K 16..23; lanes 16-31: +8.
// ---------------------------------------------------------------------------
template <int K, int NC, bool BIAS, bool RELU>
__global__ void wmma_gemm_kernel(const float* __restrict__ A,
                                 const _Float16* __restrict__ Bpack,
                                 const float* __restrict__ bias,
                                 float* __restrict__ C) {
  constexpr int NQ  = K / 32;
  constexpr int LDA = K + 4;  // pad to break LDS bank conflicts
  __shared__ __align__(16) float As[16 * LDA];

  const int m0  = blockIdx.x * 16;
  const int tid = threadIdx.x;

#if USE_ASYNC_LDS
  {
    constexpr int CPR = K / 4;       // 16-byte chunks per row
    constexpr int CH  = 16 * CPR;    // chunks per tile
    for (int ch = tid; ch < CH; ch += blockDim.x) {
      int r = ch / CPR, c = ch % CPR;
      const float* g = A + (size_t)(m0 + r) * K + c * 4;
      float*       l = As + r * LDA + c * 4;
      __builtin_amdgcn_global_load_async_to_lds_b128(
          (GLOBAL_AS v4i*)(void*)g,
          (LDS_AS v4i*)(void*)l,
          /*imm offset=*/0, /*cpol=*/0);
    }
#  if __has_builtin(__builtin_amdgcn_s_wait_asynccnt)
    __builtin_amdgcn_s_wait_asynccnt(0);
#  else
    asm volatile("s_wait_asynccnt 0x0" ::: "memory");
#  endif
  }
#else
  for (int idx = tid; idx < 16 * K; idx += blockDim.x) {
    int r = idx / K, c = idx % K;
    As[r * LDA + c] = A[(size_t)(m0 + r) * K + c];
  }
#endif
  __syncthreads();

  const int wave = tid >> 5;
  const int lane = tid & 31;
  const int g    = lane >> 4;
  const int mrow = lane & 15;

  v8f acc = {};
  #pragma unroll
  for (int q = 0; q < NQ; ++q) {
    v16h a, b;
    const int kbase = q * 32 + 8 * g;
    #pragma unroll
    for (int h = 0; h < 8; ++h) {
      a[h]     = (_Float16)As[mrow * LDA + kbase + h];
      a[h + 8] = (_Float16)As[mrow * LDA + kbase + 16 + h];
    }
    b = *(const v16h*)(Bpack + (((size_t)wave * NQ + q) * 32 + lane) * 16);
    acc = __builtin_amdgcn_wmma_f32_16x16x32_f16(
        /*neg_a=*/false, a, /*neg_b=*/false, b,
        /*c_mod=*/(short)0, acc, /*reuse_a=*/false, /*reuse_b=*/false);
  }

  const int n = wave * 16 + (lane & 15);
  #pragma unroll
  for (int r = 0; r < 8; ++r) {
    int m = m0 + r + 8 * g;
    float v = acc[r];
    if (BIAS) v += bias[n];
    if (RELU) v = fmaxf(v, 0.f);
    C[(size_t)m * NC + n] = v;
  }
}

// ---------------------------------------------------------------------------
// Graph kernels
// ---------------------------------------------------------------------------
__global__ void zero_kernel(float* __restrict__ p, size_t n) {
  for (size_t i = blockIdx.x * (size_t)blockDim.x + threadIdx.x; i < n;
       i += (size_t)gridDim.x * blockDim.x)
    p[i] = 0.f;
}

__global__ void deg_init_kernel(float* __restrict__ deg, int N) {
  int i = blockIdx.x * blockDim.x + threadIdx.x;
  if (i < N) deg[i] = 1.0f;  // self-loop
}

__global__ void deg_edge_kernel(const int* __restrict__ dst, float* __restrict__ deg, int E) {
  int e = blockIdx.x * blockDim.x + threadIdx.x;
  if (e < E) atomicAdd(&deg[dst[e]], 1.0f);
}

__global__ void dinv_kernel(float* __restrict__ deg, int N) {
  int i = blockIdx.x * blockDim.x + threadIdx.x;
  if (i < N) {
    float d = deg[i];
    deg[i] = d > 0.f ? rsqrtf(d) : 0.f;
  }
}

// one wave per work item (E edges then N self-loops): acc[d] += xw[s]*norm
__global__ void gcn_scatter_kernel(const int* __restrict__ src, const int* __restrict__ dst,
                                   const float* __restrict__ dinv, const float* __restrict__ xw,
                                   float* __restrict__ acc, int E, int N) {
  int w    = (int)((blockIdx.x * (size_t)blockDim.x + threadIdx.x) >> 5);
  int lane = threadIdx.x & 31;
  if (w >= E + N) return;
  int s, d;
  if (w < E) { s = src[w]; d = dst[w]; } else { s = d = w - E; }
  float nrm = dinv[s] * dinv[d];
  const float* xs = xw + (size_t)s * HID;
  float* ad = acc + (size_t)d * HID;
  atomicAdd(&ad[lane],      xs[lane]      * nrm);
  atomicAdd(&ad[lane + 32], xs[lane + 32] * nrm);
}

__global__ void h1_fin_kernel(const float* __restrict__ acc, const float* __restrict__ b,
                              float* __restrict__ h1, int total) {
  int i = blockIdx.x * blockDim.x + threadIdx.x;
  if (i < total) h1[i] = fmaxf(acc[i] + b[i & (HID - 1)], 0.f);
}

__device__ __forceinline__ float leaky(float v) { return v > 0.f ? v : NEG_SLOPE * v; }

// wave per node: a_s/a_d dot products + seed segment-max with the self edge
__global__ void att_kernel(const float* __restrict__ z, const float* __restrict__ att_src,
                           const float* __restrict__ att_dst, float* __restrict__ a_s,
                           float* __restrict__ a_d, float* __restrict__ mmax, int N) {
  int n    = (int)((blockIdx.x * (size_t)blockDim.x + threadIdx.x) >> 5);
  int lane = threadIdx.x & 31;
  if (n >= N) return;
  const float* zr = z + (size_t)n * (HEADS * HID);
  #pragma unroll
  for (int h = 0; h < HEADS; ++h) {
    float ps = zr[h * HID + lane] * att_src[h * HID + lane] +
               zr[h * HID + 32 + lane] * att_src[h * HID + 32 + lane];
    float pd = zr[h * HID + lane] * att_dst[h * HID + lane] +
               zr[h * HID + 32 + lane] * att_dst[h * HID + 32 + lane];
    #pragma unroll
    for (int off = 16; off; off >>= 1) {
      ps += __shfl_xor(ps, off, 32);
      pd += __shfl_xor(pd, off, 32);
    }
    if (lane == 0) {
      a_s[n * HEADS + h] = ps;
      a_d[n * HEADS + h] = pd;
      mmax[n * HEADS + h] = leaky(ps + pd);  // self edge seeds the max
    }
  }
}

// sign-split float atomic max (works for mixed-sign concurrent updates)
__device__ __forceinline__ void atomicMaxF(float* addr, float v) {
  if (v >= 0.f) atomicMax((int*)addr, __float_as_int(v));
  else          atomicMin((unsigned int*)addr, (unsigned int)__float_as_int(v));
}

__global__ void gat_max_kernel(const int* __restrict__ src, const int* __restrict__ dst,
                               const float* __restrict__ a_s, const float* __restrict__ a_d,
                               float* __restrict__ mmax, int E) {
  int e = blockIdx.x * blockDim.x + threadIdx.x;
  if (e >= E) return;
  int s = src[e], d = dst[e];
  #pragma unroll
  for (int h = 0; h < HEADS; ++h)
    atomicMaxF(&mmax[d * HEADS + h], leaky(a_s[s * HEADS + h] + a_d[d * HEADS + h]));
}

// wave per (edge|self): denom[d,h] += ex ; agg[d,:] += z[s,:]*ex (unnormalized)
__global__ void gat_agg_kernel(const int* __restrict__ src, const int* __restrict__ dst,
                               const float* __restrict__ a_s, const float* __restrict__ a_d,
                               const float* __restrict__ mmax, const float* __restrict__ z,
                               float* __restrict__ agg, float* __restrict__ denom,
                               int E, int N) {
  int w    = (int)((blockIdx.x * (size_t)blockDim.x + threadIdx.x) >> 5);
  int lane = threadIdx.x & 31;
  if (w >= E + N) return;
  int s, d;
  if (w < E) { s = src[w]; d = dst[w]; } else { s = d = w - E; }
  if (lane < HEADS) {
    float v = leaky(a_s[s * HEADS + lane] + a_d[d * HEADS + lane]);
    atomicAdd(&denom[d * HEADS + lane], __expf(v - mmax[d * HEADS + lane]));
  }
  const float* zs = z + (size_t)s * (HEADS * HID);
  float* ag = agg + (size_t)d * (HEADS * HID);
  #pragma unroll
  for (int it = 0; it < 8; ++it) {
    int h = it >> 1;  // uniform per iteration
    float v  = leaky(a_s[s * HEADS + h] + a_d[d * HEADS + h]);
    float ex = __expf(v - mmax[d * HEADS + h]);
    int c = it * 32 + lane;
    atomicAdd(&ag[c], zs[c] * ex);
  }
}

// wave per node: h2 = relu(mean_h(agg/denom) + b_gat); h = h1 + h2 + hpatch;
// LayerNorm over 64 channels via shuffle reductions.
__global__ void final_kernel(const float* __restrict__ h1, const float* __restrict__ agg,
                             const float* __restrict__ denom, const float* __restrict__ hpatch,
                             const float* __restrict__ b_gat, const float* __restrict__ gamma,
                             const float* __restrict__ beta, float* __restrict__ out, int N) {
  int n    = (int)((blockIdx.x * (size_t)blockDim.x + threadIdx.x) >> 5);
  int lane = threadIdx.x & 31;
  if (n >= N) return;
  float invd[HEADS];
  #pragma unroll
  for (int h = 0; h < HEADS; ++h) invd[h] = 1.f / denom[n * HEADS + h];
  const float* ag = agg + (size_t)n * (HEADS * HID);
  float hv[2];
  #pragma unroll
  for (int j = 0; j < 2; ++j) {
    int c = lane + 32 * j;
    float a = 0.f;
    #pragma unroll
    for (int h = 0; h < HEADS; ++h) a += ag[h * HID + c] * invd[h];
    float h2 = fmaxf(a * (1.0f / HEADS) + b_gat[c], 0.f);
    hv[j] = h1[(size_t)n * HID + c] + h2 + hpatch[(size_t)n * HID + c];
  }
  float sum = hv[0] + hv[1];
  #pragma unroll
  for (int off = 16; off; off >>= 1) sum += __shfl_xor(sum, off, 32);
  float mu = sum * (1.f / HID);
  float v0 = hv[0] - mu, v1 = hv[1] - mu;
  float vs = v0 * v0 + v1 * v1;
  #pragma unroll
  for (int off = 16; off; off >>= 1) vs += __shfl_xor(vs, off, 32);
  float rstd = rsqrtf(vs * (1.f / HID) + LN_EPS);
  #pragma unroll
  for (int j = 0; j < 2; ++j) {
    int c = lane + 32 * j;
    out[(size_t)n * HID + c] = (hv[j] - mu) * rstd * gamma[c] + beta[c];
  }
}

// ---------------------------------------------------------------------------
extern "C" void kernel_launch(void* const* d_in, const int* in_sizes, int n_in,
                              void* d_out, int out_size, void* d_ws, size_t ws_size,
                              hipStream_t stream) {
  const float* x       = (const float*)d_in[0];
  const int*   eidx    = (const int*)d_in[1];
  const float* W_gcn   = (const float*)d_in[2];
  const float* b_gcn   = (const float*)d_in[3];
  const float* W_gat   = (const float*)d_in[4];
  const float* att_src = (const float*)d_in[5];
  const float* att_dst = (const float*)d_in[6];
  const float* b_gat   = (const float*)d_in[7];
  const float* W_patch = (const float*)d_in[8];
  const float* b_patch = (const float*)d_in[9];
  const float* gamma   = (const float*)d_in[10];
  const float* beta    = (const float*)d_in[11];

  const int N = in_sizes[0] / 128;   // 50000 (multiple of 16)
  const int E = in_sizes[1] / 2;     // 400000
  const int* src = eidx;
  const int* dst = eidx + E;

  // ---- workspace layout (floats) ----
  float* f = (float*)d_ws;
  _Float16* pack1 = (_Float16*)f;            //  4*4*32*16 =  8192 halves (W_gcn 128x64)
  _Float16* pack2 = pack1 + 8192;            // 16*2*32*16 = 16384 halves (W_gat 64x256)
  _Float16* pack3 = pack2 + 16384;           //  4*2*32*16 =  4096 halves (W_patch 64x64)
  size_t off = 14336;                        // 28672 halves = 14336 floats
  float* deg   = f + off; off += (size_t)N;
  float* a_s   = f + off; off += (size_t)4 * N;
  float* a_d   = f + off; off += (size_t)4 * N;
  float* mmax  = f + off; off += (size_t)4 * N;
  float* xw    = f + off; off += (size_t)64 * N;
  float* h1    = f + off; off += (size_t)64 * N;
  float* z     = f + off; off += (size_t)256 * N;
  float* hpat  = f + off; off += (size_t)64 * N;
  float* acc   = f + off; off += (size_t)64 * N;    // --- zeroed region start
  float* agg   = f + off; off += (size_t)256 * N;
  float* denom = f + off; off += (size_t)4 * N;     // --- zeroed region end

  const int nWaveItems = E + N;                       // edges + self loops
  const int gWave = (nWaveItems + 7) / 8;             // 8 waves / 256-thread block
  const int gNode = (N + 7) / 8;

  // 1) pack weights into WMMA B-fragment layout (f16)
  pack_b_kernel<<<(4 * 4 * 32 + 255) / 256, 256, 0, stream>>>(W_gcn, pack1, 128, 64);
  pack_b_kernel<<<(16 * 2 * 32 + 255) / 256, 256, 0, stream>>>(W_gat, pack2, 64, 256);
  pack_b_kernel<<<(4 * 2 * 32 + 255) / 256, 256, 0, stream>>>(W_patch, pack3, 64, 64);

  // 2) zero accumulators (acc | agg | denom contiguous)
  zero_kernel<<<4096, 256, 0, stream>>>(acc, (size_t)N * (64 + 256 + 4));

  // 3) degrees -> dinv (in place)
  deg_init_kernel<<<(N + 255) / 256, 256, 0, stream>>>(deg, N);
  deg_edge_kernel<<<(E + 255) / 256, 256, 0, stream>>>(dst, deg, E);
  dinv_kernel<<<(N + 255) / 256, 256, 0, stream>>>(deg, N);

  // 4) xw = x @ W_gcn  (WMMA, async-LDS staged)
  wmma_gemm_kernel<128, 64, false, false><<<N / 16, 128, 0, stream>>>(x, pack1, nullptr, xw);

  // 5) GCN scatter + finalize h1
  gcn_scatter_kernel<<<gWave, 256, 0, stream>>>(src, dst, deg, xw, acc, E, N);
  h1_fin_kernel<<<(N * 64 + 255) / 256, 256, 0, stream>>>(acc, b_gcn, h1, N * 64);

  // 6) z = h1 @ W_gat  (WMMA)
  wmma_gemm_kernel<64, 256, false, false><<<N / 16, 512, 0, stream>>>(h1, pack2, nullptr, z);

  // 7) GAT attention: dots, segment max, exp-sum + unnormalized aggregation
  att_kernel<<<gNode, 256, 0, stream>>>(z, att_src, att_dst, a_s, a_d, mmax, N);
  gat_max_kernel<<<(E + 255) / 256, 256, 0, stream>>>(src, dst, a_s, a_d, mmax, E);
  gat_agg_kernel<<<gWave, 256, 0, stream>>>(src, dst, a_s, a_d, mmax, z, agg, denom, E, N);

  // 8) hpatch = h1 @ W_patch + b_patch  (WMMA)
  wmma_gemm_kernel<64, 64, true, false><<<N / 16, 128, 0, stream>>>(h1, pack3, b_patch, hpat);

  // 9) fuse: head-mean + bias + relu, residual sum, LayerNorm
  final_kernel<<<gNode, 256, 0, stream>>>(h1, agg, denom, hpat, b_gat, gamma, beta,
                                          (float*)d_out, N);
}
